// Block_74861279969699
// MI455X (gfx1250) — compile-verified, matrix operand order checked by reading, not run on hardware
//
#include <hip/hip_runtime.h>

// ---------------------------------------------------------------------------
// Types for CDNA5 WMMA (gfx1250, wave32)
// ---------------------------------------------------------------------------
typedef __bf16 bf16_t;
typedef __attribute__((ext_vector_type(16))) __bf16 bf16x16;
typedef __attribute__((ext_vector_type(8)))  __bf16 bf16x8;
typedef __attribute__((ext_vector_type(8)))  float  v8f;

union Frag16 {
    bf16x16 v;
    bf16x8  h[2];
};

__device__ inline bf16x8 load8(const bf16_t* p) {
    return *reinterpret_cast<const bf16x8*>(p);
}

__device__ inline v8f v8f_zero() {
    v8f z = {0.f, 0.f, 0.f, 0.f, 0.f, 0.f, 0.f, 0.f};
    return z;
}

__device__ inline v8f wmma_bf16(const Frag16& a, const Frag16& b, v8f c) {
    // D = A(16x32 bf16) * B(32x16 bf16) + C(16x16 f32)
    return __builtin_amdgcn_wmma_f32_16x16x32_bf16(
        /*neg_a=*/false, a.v, /*neg_b=*/false, b.v,
        /*c_mod=*/(short)0, c, /*reuse_a=*/false, /*reuse_b=*/false);
}

// ---------------------------------------------------------------------------
// Constants
// ---------------------------------------------------------------------------
#define DIMC   1024
#define NHEAD  16
#define HDIM   64
#define SEQN   1024
#define BATCH  4
#define MROWS  4096      // BATCH * SEQN
#define HIDDEN 4096

// ---------------------------------------------------------------------------
// f32 -> bf16 conversion (weights, once per call)
// ---------------------------------------------------------------------------
__global__ void cvt_f32_bf16_kernel(const float* __restrict__ src,
                                    bf16_t* __restrict__ dst, int n) {
    int i = blockIdx.x * blockDim.x + threadIdx.x;
    if (i < n) dst[i] = (bf16_t)src[i];
}

// ---------------------------------------------------------------------------
// LayerNorm: one row (1024 f32) per workgroup (8 waves), bf16 output
// ---------------------------------------------------------------------------
__global__ void layernorm_bf16_kernel(const float* __restrict__ x,
                                      const float* __restrict__ w,
                                      const float* __restrict__ b,
                                      bf16_t* __restrict__ out, int cols) {
    int row = blockIdx.x;
    const float* xr = x + (size_t)row * cols;
    float s1 = 0.f, s2 = 0.f;
    for (int c = threadIdx.x; c < cols; c += blockDim.x) {
        float v = xr[c];
        s1 += v;
        s2 += v * v;
    }
#pragma unroll
    for (int off = 16; off > 0; off >>= 1) {
        s1 += __shfl_xor(s1, off, 32);
        s2 += __shfl_xor(s2, off, 32);
    }
    __shared__ float r1[8], r2[8];
    int wid = threadIdx.x >> 5;
    int lane = threadIdx.x & 31;
    if (lane == 0) { r1[wid] = s1; r2[wid] = s2; }
    __syncthreads();
    if (threadIdx.x == 0) {
        float a = 0.f, bb = 0.f;
        for (int i = 0; i < 8; i++) { a += r1[i]; bb += r2[i]; }
        r1[0] = a; r2[0] = bb;
    }
    __syncthreads();
    float mu  = r1[0] / cols;
    float var = r2[0] / cols - mu * mu;
    float rs  = rsqrtf(var + 1e-5f);
    for (int c = threadIdx.x; c < cols; c += blockDim.x) {
        out[(size_t)row * cols + c] = (bf16_t)((xr[c] - mu) * rs * w[c] + b[c]);
    }
}

// ---------------------------------------------------------------------------
// Core: one wave computes a 64(M) x 32(N) output tile of A[MxK] * W[NxK]^T.
// A row-major [M,K] bf16, W row-major [N,K] bf16 (so B[k][n] = W[n][k]).
// acc[i][j] = C-frag for M-subtile i (0..3) and N-subtile j (0..1).
// ---------------------------------------------------------------------------
__device__ inline void wave_gemm_64x32(const bf16_t* __restrict__ A,
                                       const bf16_t* __restrict__ W,
                                       int K, int m0, int n0, v8f acc[4][2]) {
    const int lane = threadIdx.x & 31;
    const int hf   = lane >> 4;     // lane half selects K sub-stripe
    const int lr   = lane & 15;     // M row (A) / N col (B) within 16
#pragma unroll
    for (int i = 0; i < 4; i++)
#pragma unroll
        for (int j = 0; j < 2; j++)
            acc[i][j] = v8f_zero();

    for (int kk = 0; kk < K; kk += 32) {
        Frag16 a[4], b[2];
#pragma unroll
        for (int i = 0; i < 4; i++) {
            const bf16_t* p = A + (size_t)(m0 + i * 16 + lr) * K + kk + hf * 8;
            a[i].h[0] = load8(p);        // K = hf*8 .. hf*8+7
            a[i].h[1] = load8(p + 16);   // K = 16+hf*8 .. 16+hf*8+7
        }
#pragma unroll
        for (int j = 0; j < 2; j++) {
            const bf16_t* p = W + (size_t)(n0 + j * 16 + lr) * K + kk + hf * 8;
            b[j].h[0] = load8(p);
            b[j].h[1] = load8(p + 16);
        }
#pragma unroll
        for (int i = 0; i < 4; i++)
#pragma unroll
            for (int j = 0; j < 2; j++)
                acc[i][j] = wmma_bf16(a[i], b[j], acc[i][j]);
    }
}

// ---------------------------------------------------------------------------
// QKV GEMM: [4096,1024] x [3072,1024]^T, epilogue scatters into
//   Q [B,H,N,64] (scaled by hd^-0.5, +q_bias), K [B,H,N,64],
//   V^T [B,H,64,N] (+v_bias)  -- V transposed so P*V B-frags load contiguous.
// grid (64, 12), block 256 (8 waves, each 64x32)
// ---------------------------------------------------------------------------
__global__ void gemm_qkv_kernel(const bf16_t* __restrict__ A,
                                const bf16_t* __restrict__ W,
                                const float* __restrict__ q_bias,
                                const float* __restrict__ v_bias,
                                bf16_t* __restrict__ qo,
                                bf16_t* __restrict__ ko,
                                bf16_t* __restrict__ vto) {
    int wv = threadIdx.x >> 5;
    int m0 = blockIdx.x * 64;
    int n0 = blockIdx.y * 256 + wv * 32;
    v8f acc[4][2];
    wave_gemm_64x32(A, W, DIMC, m0, n0, acc);

    int lane = threadIdx.x & 31, hf = lane >> 4, lr = lane & 15;
#pragma unroll
    for (int i = 0; i < 4; i++)
#pragma unroll
        for (int j = 0; j < 2; j++) {
            int col    = n0 + j * 16 + lr;      // 0..3071
            int sec    = col >> 10;             // 0=q 1=k 2=v
            int within = col & 1023;
            int hh     = within >> 6;
            int d      = within & 63;
            float bias = (sec == 0) ? q_bias[within]
                       : (sec == 2) ? v_bias[within] : 0.f;
#pragma unroll
            for (int r = 0; r < 8; r++) {
                int row = m0 + i * 16 + 8 * hf + r;  // global token row
                int bi  = row >> 10;
                int n   = row & 1023;
                float val = acc[i][j][r] + bias;
                size_t bh = (size_t)bi * NHEAD + hh;
                if (sec == 0)
                    qo[(bh * SEQN + n) * HDIM + d] = (bf16_t)(val * 0.125f);
                else if (sec == 1)
                    ko[(bh * SEQN + n) * HDIM + d] = (bf16_t)val;
                else
                    vto[(bh * HDIM + d) * SEQN + n] = (bf16_t)val;
            }
        }
}

// ---------------------------------------------------------------------------
// Flash attention: one wave per (b, h, 16-query tile).
// S(16x32) = Q(16x64) * K_tile^T via 2 WMMAs per key-subtile; + rel_pos_bias,
// key-padding mask; online softmax; O(16x64) += P(16x32)*V(32x64) via 4 WMMAs.
// P transposed C-layout -> A-layout through LDS.
// grid 4096, block 32.
// ---------------------------------------------------------------------------
__global__ void attention_kernel(const bf16_t* __restrict__ Q,
                                 const bf16_t* __restrict__ Km,
                                 const bf16_t* __restrict__ Vt,
                                 const float* __restrict__ rel,
                                 const int* __restrict__ amask,
                                 bf16_t* __restrict__ out) {
    int blk = blockIdx.x;
    int qt  = blk & 63;          // SEQN/16 = 64 query tiles
    int bh  = blk >> 6;
    int h   = bh & (NHEAD - 1);
    int bi  = bh >> 4;
    int qbase = qt * 16;

    int lane = threadIdx.x & 31;
    int hf   = lane >> 4;
    int lr   = lane & 15;

    const bf16_t* Qb = Q + ((size_t)bh * SEQN + qbase) * HDIM;
    const bf16_t* Kb = Km + (size_t)bh * SEQN * HDIM;
    const bf16_t* Vb = Vt + (size_t)bh * HDIM * SEQN;
    const float*  Rb = rel + (size_t)h * SEQN * SEQN;
    const int*    Mb = amask + (size_t)bi * SEQN;

    // Q A-frags for the two 32-wide K(=d) halves
    Frag16 aq[2];
#pragma unroll
    for (int s = 0; s < 2; s++) {
        const bf16_t* p = Qb + (size_t)lr * HDIM + s * 32 + hf * 8;
        aq[s].h[0] = load8(p);
        aq[s].h[1] = load8(p + 16);
    }

    float mrow[8], lrow[8];
    v8f o[4];
#pragma unroll
    for (int r = 0; r < 8; r++) { mrow[r] = -1e30f; lrow[r] = 0.f; }
#pragma unroll
    for (int j = 0; j < 4; j++) o[j] = v8f_zero();

    __shared__ float sP[16][33];   // +1 pad vs banks

    for (int kk = 0; kk < SEQN; kk += 32) {
        // ---- scores for two 16-key subtiles
        v8f s[2];
#pragma unroll
        for (int t = 0; t < 2; t++) {
            int key = kk + t * 16 + lr;
            const bf16_t* kp = Kb + (size_t)key * HDIM + hf * 8;
            Frag16 b0, b1;
            b0.h[0] = load8(kp);       b0.h[1] = load8(kp + 16);
            b1.h[0] = load8(kp + 32);  b1.h[1] = load8(kp + 48);
            s[t] = wmma_bf16(aq[0], b0, v8f_zero());
            s[t] = wmma_bf16(aq[1], b1, s[t]);
        }
        // ---- bias + key-padding mask
        float sv[2][8];
#pragma unroll
        for (int t = 0; t < 2; t++) {
            int key = kk + t * 16 + lr;
            int mk  = Mb[key];
            const float* bp = Rb + key;
#pragma unroll
            for (int r = 0; r < 8; r++) {
                int qrow = qbase + 8 * hf + r;
                float v  = s[t][r] + bp[(size_t)qrow * SEQN];
                sv[t][r] = mk ? v : -1e30f;
            }
        }
        // ---- online softmax: row max across the 16-lane group
        float mnew[8], scale[8];
#pragma unroll
        for (int r = 0; r < 8; r++) {
            float v = fmaxf(sv[0][r], sv[1][r]);
#pragma unroll
            for (int off = 1; off < 16; off <<= 1)
                v = fmaxf(v, __shfl_xor(v, off, 32));
            mnew[r]  = fmaxf(mrow[r], v);
            scale[r] = __expf(mrow[r] - mnew[r]);
        }
        float ps[2][8];
#pragma unroll
        for (int t = 0; t < 2; t++)
#pragma unroll
            for (int r = 0; r < 8; r++)
                ps[t][r] = (sv[t][r] > -1e29f) ? __expf(sv[t][r] - mnew[r]) : 0.f;
#pragma unroll
        for (int r = 0; r < 8; r++) {
            float srow = ps[0][r] + ps[1][r];
#pragma unroll
            for (int off = 1; off < 16; off <<= 1)
                srow += __shfl_xor(srow, off, 32);
            lrow[r] = lrow[r] * scale[r] + srow;
            mrow[r] = mnew[r];
        }
#pragma unroll
        for (int j = 0; j < 4; j++)
#pragma unroll
            for (int r = 0; r < 8; r++)
                o[j][r] = o[j][r] * scale[r];

        // ---- P: C-layout -> A-layout via LDS transpose, cast to bf16
#pragma unroll
        for (int t = 0; t < 2; t++)
#pragma unroll
            for (int r = 0; r < 8; r++)
                sP[8 * hf + r][t * 16 + lr] = ps[t][r];
        __syncthreads();
        Frag16 ap;
#pragma unroll
        for (int i = 0; i < 8; i++) {
            ap.v[i]     = (bf16_t)sP[lr][hf * 8 + i];
            ap.v[i + 8] = (bf16_t)sP[lr][16 + hf * 8 + i];
        }
        __syncthreads();

        // ---- O += P * V  (V^T layout => contiguous B-frag loads)
#pragma unroll
        for (int j = 0; j < 4; j++) {
            const bf16_t* vp = Vb + (size_t)(j * 16 + lr) * SEQN + kk + hf * 8;
            Frag16 bv;
            bv.h[0] = load8(vp);
            bv.h[1] = load8(vp + 16);
            o[j] = wmma_bf16(ap, bv, o[j]);
        }
    }

    // ---- normalize and write [B,N, H*64] bf16
#pragma unroll
    for (int r = 0; r < 8; r++) {
        float inv = (lrow[r] > 0.f) ? 1.f / lrow[r] : 0.f;
        int qrow = qbase + 8 * hf + r;
        size_t rowoff = ((size_t)bi * SEQN + qrow) * DIMC + h * HDIM;
#pragma unroll
        for (int j = 0; j < 4; j++)
            out[rowoff + j * 16 + lr] = (bf16_t)(o[j][r] * inv);
    }
}

// ---------------------------------------------------------------------------
// Proj GEMM + bias + residual -> f32.  grid (64,4), block 256
// ---------------------------------------------------------------------------
__global__ void gemm_proj_kernel(const bf16_t* __restrict__ A,
                                 const bf16_t* __restrict__ W,
                                 const float* __restrict__ bias,
                                 const float* __restrict__ resid,
                                 float* __restrict__ out) {
    int wv = threadIdx.x >> 5;
    int m0 = blockIdx.x * 64;
    int n0 = blockIdx.y * 256 + wv * 32;
    v8f acc[4][2];
    wave_gemm_64x32(A, W, DIMC, m0, n0, acc);
    int lane = threadIdx.x & 31, hf = lane >> 4, lr = lane & 15;
#pragma unroll
    for (int i = 0; i < 4; i++)
#pragma unroll
        for (int j = 0; j < 2; j++) {
            int col = n0 + j * 16 + lr;
            float bv = bias[col];
#pragma unroll
            for (int r = 0; r < 8; r++) {
                size_t row = (size_t)(m0 + i * 16 + 8 * hf + r);
                out[row * DIMC + col] = acc[i][j][r] + bv + resid[row * DIMC + col];
            }
        }
}

// ---------------------------------------------------------------------------
// MLP1 GEMM + bias + exact GELU -> bf16.  grid (64,16), block 256
// ---------------------------------------------------------------------------
__global__ void gemm_mlp1_kernel(const bf16_t* __restrict__ A,
                                 const bf16_t* __restrict__ W,
                                 const float* __restrict__ bias,
                                 bf16_t* __restrict__ out) {
    int wv = threadIdx.x >> 5;
    int m0 = blockIdx.x * 64;
    int n0 = blockIdx.y * 256 + wv * 32;
    v8f acc[4][2];
    wave_gemm_64x32(A, W, DIMC, m0, n0, acc);
    int lane = threadIdx.x & 31, hf = lane >> 4, lr = lane & 15;
#pragma unroll
    for (int i = 0; i < 4; i++)
#pragma unroll
        for (int j = 0; j < 2; j++) {
            int col = n0 + j * 16 + lr;
            float bv = bias[col];
#pragma unroll
            for (int r = 0; r < 8; r++) {
                size_t row = (size_t)(m0 + i * 16 + 8 * hf + r);
                float v = acc[i][j][r] + bv;
                float g = 0.5f * v * (1.f + erff(v * 0.70710678118654752f));
                out[row * HIDDEN + col] = (bf16_t)g;
            }
        }
}

// ---------------------------------------------------------------------------
// MLP2 GEMM + bias + residual -> f32 final output.  grid (64,4), block 256
// ---------------------------------------------------------------------------
__global__ void gemm_mlp2_kernel(const bf16_t* __restrict__ A,
                                 const bf16_t* __restrict__ W,
                                 const float* __restrict__ bias,
                                 const float* __restrict__ resid,
                                 float* __restrict__ out) {
    int wv = threadIdx.x >> 5;
    int m0 = blockIdx.x * 64;
    int n0 = blockIdx.y * 256 + wv * 32;
    v8f acc[4][2];
    wave_gemm_64x32(A, W, HIDDEN, m0, n0, acc);
    int lane = threadIdx.x & 31, hf = lane >> 4, lr = lane & 15;
#pragma unroll
    for (int i = 0; i < 4; i++)
#pragma unroll
        for (int j = 0; j < 2; j++) {
            int col = n0 + j * 16 + lr;
            float bv = bias[col];
#pragma unroll
            for (int r = 0; r < 8; r++) {
                size_t row = (size_t)(m0 + i * 16 + 8 * hf + r);
                out[row * DIMC + col] = acc[i][j][r] + bv + resid[row * DIMC + col];
            }
        }
}

// ---------------------------------------------------------------------------
// Host driver
// ---------------------------------------------------------------------------
extern "C" void kernel_launch(void* const* d_in, const int* in_sizes, int n_in,
                              void* d_out, int out_size, void* d_ws, size_t ws_size,
                              hipStream_t stream) {
    const float* x      = (const float*)d_in[0];
    const float* rel    = (const float*)d_in[1];
    const int*   amask  = (const int*)  d_in[2];
    const float* n1w    = (const float*)d_in[3];
    const float* n1b    = (const float*)d_in[4];
    const float* qkv_w  = (const float*)d_in[5];
    const float* q_bias = (const float*)d_in[6];
    const float* v_bias = (const float*)d_in[7];
    const float* proj_w = (const float*)d_in[8];
    const float* proj_b = (const float*)d_in[9];
    const float* n2w    = (const float*)d_in[10];
    const float* n2b    = (const float*)d_in[11];
    const float* w1     = (const float*)d_in[12];
    const float* b1     = (const float*)d_in[13];
    const float* w2     = (const float*)d_in[14];
    const float* b2     = (const float*)d_in[15];
    float* out = (float*)d_out;
    (void)in_sizes; (void)n_in; (void)out_size; (void)ws_size;

    char* ws = (char*)d_ws;
    size_t off = 0;
    auto alloc = [&](size_t bytes) -> char* {
        char* p = ws + off;
        off = (off + bytes + 255) & ~(size_t)255;
        return p;
    };
    bf16_t* ln1b   = (bf16_t*)alloc((size_t)MROWS * DIMC * 2);
    bf16_t* wqkvb  = (bf16_t*)alloc((size_t)3 * DIMC * DIMC * 2);
    bf16_t* qb     = (bf16_t*)alloc((size_t)MROWS * DIMC * 2);   // [B,H,N,64]
    bf16_t* kb     = (bf16_t*)alloc((size_t)MROWS * DIMC * 2);   // [B,H,N,64]
    bf16_t* vtb    = (bf16_t*)alloc((size_t)MROWS * DIMC * 2);   // [B,H,64,N]
    bf16_t* attnb  = (bf16_t*)alloc((size_t)MROWS * DIMC * 2);   // [B*N, C]
    bf16_t* wprojb = (bf16_t*)alloc((size_t)DIMC * DIMC * 2);
    float*  x1     = (float*) alloc((size_t)MROWS * DIMC * 4);
    bf16_t* ln2b   = (bf16_t*)alloc((size_t)MROWS * DIMC * 2);
    bf16_t* w1b    = (bf16_t*)alloc((size_t)HIDDEN * DIMC * 2);
    bf16_t* hb     = (bf16_t*)alloc((size_t)MROWS * HIDDEN * 2);
    bf16_t* w2b    = (bf16_t*)alloc((size_t)DIMC * HIDDEN * 2);

    // weight conversions
    cvt_f32_bf16_kernel<<<(3 * DIMC * DIMC + 255) / 256, 256, 0, stream>>>(qkv_w, wqkvb, 3 * DIMC * DIMC);
    cvt_f32_bf16_kernel<<<(DIMC * DIMC + 255) / 256, 256, 0, stream>>>(proj_w, wprojb, DIMC * DIMC);
    cvt_f32_bf16_kernel<<<(HIDDEN * DIMC + 255) / 256, 256, 0, stream>>>(w1, w1b, HIDDEN * DIMC);
    cvt_f32_bf16_kernel<<<(DIMC * HIDDEN + 255) / 256, 256, 0, stream>>>(w2, w2b, DIMC * HIDDEN);

    // x -> LN1 -> bf16
    layernorm_bf16_kernel<<<MROWS, 256, 0, stream>>>(x, n1w, n1b, ln1b, DIMC);

    // QKV projection + head scatter
    gemm_qkv_kernel<<<dim3(MROWS / 64, (3 * DIMC) / 256), 256, 0, stream>>>(
        ln1b, wqkvb, q_bias, v_bias, qb, kb, vtb);

    // flash attention per (b, h, 16-query tile)
    attention_kernel<<<BATCH * NHEAD * (SEQN / 16), 32, 0, stream>>>(
        qb, kb, vtb, rel, amask, attnb);

    // proj + residual -> x1 (f32)
    gemm_proj_kernel<<<dim3(MROWS / 64, DIMC / 256), 256, 0, stream>>>(
        attnb, wprojb, proj_b, x, x1);

    // LN2 -> bf16
    layernorm_bf16_kernel<<<MROWS, 256, 0, stream>>>(x1, n2w, n2b, ln2b, DIMC);

    // MLP
    gemm_mlp1_kernel<<<dim3(MROWS / 64, HIDDEN / 256), 256, 0, stream>>>(
        ln2b, w1b, b1, hb);
    gemm_mlp2_kernel<<<dim3(MROWS / 64, DIMC / 256), 256, 0, stream>>>(
        hb, w2b, b2, x1, out);
}
